// GraphAttn_70970039599321
// MI455X (gfx1250) — matrix operand order repaired and checked by reading, not compiled, not run on hardware
//
#include <hip/hip_runtime.h>
#include <hip/hip_bf16.h>
#include <stdint.h>

typedef __attribute__((ext_vector_type(16))) _Float16 v16h;
typedef __attribute__((ext_vector_type(8)))  _Float16 v8h;
typedef __attribute__((ext_vector_type(8)))  float    v8f;
typedef __attribute__((ext_vector_type(4)))  float    v4f;
typedef int v4i_gcc __attribute__((vector_size(16)));   // matches async-LDS builtin param

#define N_ROWS 16384
#define IN_DIM 256
#define OUT_DIM 128
#define NEG_SLOPE 0.01f

#if defined(__has_builtin)
#if __has_builtin(__builtin_amdgcn_global_load_async_to_lds_b128)
#define HAVE_ASYNC_LDS 1
#endif
#if __has_builtin(__builtin_elementwise_maximumnum)
#define VMAX(a, b) __builtin_elementwise_maximumnum(a, b)
#else
#define VMAX(a, b) __builtin_elementwise_max(a, b)
#endif
#else
#define VMAX(a, b) __builtin_elementwise_max(a, b)
#endif

// ---------------------------------------------------------------------------
// Phase 1: h = x @ W^T stored transposed as f16: hT[c * N + i].
// One block per row i; x row broadcast, W rows L2-resident. ~1.5% of FLOPs.
// ---------------------------------------------------------------------------
__global__ __launch_bounds__(OUT_DIM)
void phase1_h(const float* __restrict__ x, const float* __restrict__ W,
              _Float16* __restrict__ hT) {
    const int i = blockIdx.x;
    const int c = threadIdx.x;
    const v4f* xr = (const v4f*)(x + (size_t)i * IN_DIM);
    const v4f* wr = (const v4f*)(W + (size_t)c * IN_DIM);
    float acc = 0.f;
#pragma unroll 4
    for (int k = 0; k < IN_DIM / 4; ++k) {
        v4f xv = xr[k], wv = wr[k];
        acc += xv.x * wv.x + xv.y * wv.y + xv.z * wv.z + xv.w * wv.w;
    }
    hT[(size_t)c * N_ROWS + i] = (_Float16)acc;
}

// ---------------------------------------------------------------------------
// Phase 1b: s1[i] = h[i,:]@a1, s2[i] = h[i,:]@a2 -> stored as f16.
// ---------------------------------------------------------------------------
__global__ __launch_bounds__(256)
void phase1_s(const _Float16* __restrict__ hT,
              const float* __restrict__ a1, const float* __restrict__ a2,
              _Float16* __restrict__ s1h, _Float16* __restrict__ s2h) {
    const int i = blockIdx.x * 256 + threadIdx.x;
    float acc1 = 0.f, acc2 = 0.f;
#pragma unroll 8
    for (int c = 0; c < OUT_DIM; ++c) {
        float hv = (float)hT[(size_t)c * N_ROWS + i];
        acc1 += hv * a1[c];
        acc2 += hv * a2[c];
    }
    s1h[i] = (_Float16)acc1;
    s2h[i] = (_Float16)acc2;
}

// ---------------------------------------------------------------------------
// Phase 3: out = e @ h with e[i,j] = lrelu(s2[i]+s1[j]) synthesized in-regs.
// Block = 128 threads = 4 waves; M-tile 128 rows (32 rows/wave = 2 A frags,
// 16 C tiles = 128 accumulator VGPRs); all 128 columns.
// B (h tile, K=64 x 128 cols, f16) staged per block into double-buffered LDS
// via async global->LDS copies, shared by all 4 waves. K padded 64->72 halves
// so B-fragment reads (row stride 144B = 36 banks) are conflict-free.
// B fragments use a 2-slot rotation (slot = idx&1, reloaded 2 iters ahead)
// so the two slots have interleaved live ranges -> disjoint registers ->
// ds_load latency hides behind a full WMMA pair instead of s_wait_dscnt 0.
//   A frag (16x32 f16): lane holds row M=lane%16; K(t) = (t<8?t:t+8)+8*(lane/16)
//   B frag (32x16 f16): lane holds col N=lane%16; K(t) = (t<8?t:t+8)+8*(lane/16)
// ---------------------------------------------------------------------------
#define KSTAGE 64
#define BPAD   72   // padded K extent (halves) per column in LDS

__global__ __launch_bounds__(128)
void phase3_attn_gemm(const _Float16* __restrict__ hT,
                      const _Float16* __restrict__ s1h,
                      const _Float16* __restrict__ s2h,
                      float* __restrict__ out) {
    __shared__ _Float16 Bs[2][OUT_DIM * BPAD];   // 2 x 18 KB

    const int tid  = threadIdx.x;
    const int wave = tid >> 5;
    const int lane = tid & 31;
    const int m    = lane & 15;   // A/C row within tile; B/C column within tile
    const int g    = lane >> 4;   // lane group (selects K halves)

    const int row0 = blockIdx.x * 128 + wave * 32;
    const _Float16 s2a = s2h[row0 + m];        // rows m    (first 16-row group)
    const _Float16 s2b = s2h[row0 + 16 + m];   // rows m+16 (second group)
    const _Float16 slope = (_Float16)NEG_SLOPE;

    v8f c0[8], c1[8];
#pragma unroll
    for (int ct = 0; ct < 8; ++ct) { c0[ct] = (v8f)0.f; c1[ct] = (v8f)0.f; }

    // Cooperative stage of one K=64 x 128-col f16 tile of hT into Bs[buf].
    auto stage = [&](int buf, int kb) {
#pragma unroll
        for (int j = 0; j < 8; ++j) {
            const int id  = j * 128 + tid;   // 1024 chunks of 8 halves
            const int col = id >> 3;
            const int kc  = id & 7;
            const _Float16* src = hT + (size_t)col * N_ROWS + kb + kc * 8;
            _Float16* dst = &Bs[buf][col * BPAD + kc * 8];
#ifdef HAVE_ASYNC_LDS
            __builtin_amdgcn_global_load_async_to_lds_b128(
                (__attribute__((address_space(1))) v4i_gcc*)(uintptr_t)src,
                (__attribute__((address_space(3))) v4i_gcc*)(uint32_t)(uintptr_t)dst,
                0, 0);
#else
            *(v8h*)dst = *(const v8h*)src;
#endif
        }
    };

    stage(0, 0);  // prologue fill of buffer 0

    for (int r = 0; r < N_ROWS / KSTAGE; ++r) {
        const int kb  = r * KSTAGE;
        const int cur = r & 1;
#ifdef HAVE_ASYNC_LDS
        asm volatile("s_wait_asynccnt 0" ::: "memory");
#endif
        __syncthreads();                       // Bs[cur] ready for everyone
        if (kb + KSTAGE < N_ROWS) stage(cur ^ 1, kb + KSTAGE);

        // ---- A fragments for both K=32 sub-steps, packed f16 lrelu ----
        v16h A0[2], A1[2];
#pragma unroll
        for (int ks = 0; ks < 2; ++ks) {
            const _Float16* sp = s1h + kb + ks * 32 + 8 * g;
            v8h sa = *(const v8h*)sp;          // K = 8g+0..7    (t = 0..7)
            v8h sb = *(const v8h*)(sp + 16);   // K = 16+8g+0..7 (t = 8..15)
            v16h s1v = __builtin_shufflevector(sa, sb,
                0, 1, 2, 3, 4, 5, 6, 7, 8, 9, 10, 11, 12, 13, 14, 15);
            v16h ea = s1v + s2a;
            A0[ks] = VMAX(ea, ea * slope);
            v16h eb = s1v + s2b;
            A1[ks] = VMAX(eb, eb * slope);
        }

        // ---- flattened 16-step (2 substeps x 8 col tiles) WMMA pipeline ----
        auto loadB = [&](int idx) -> v16h {
            const int ks = idx >> 3, ct = idx & 7;
            const _Float16* bp =
                &Bs[cur][(ct * 16 + m) * BPAD + ks * 32 + 8 * g];
            v8h lo = *(const v8h*)bp;
            v8h hi = *(const v8h*)(bp + 16);
            return __builtin_shufflevector(lo, hi,
                0, 1, 2, 3, 4, 5, 6, 7, 8, 9, 10, 11, 12, 13, 14, 15);
        };

        v16h bbuf[2];
        bbuf[0] = loadB(0);
        bbuf[1] = loadB(1);
#pragma unroll
        for (int idx = 0; idx < 16; ++idx) {
            const int ks = idx >> 3, ct = idx & 7;
            const v16h buse = bbuf[idx & 1];
            c0[ct] = __builtin_amdgcn_wmma_f32_16x16x32_f16(
                false, A0[ks], false, buse, (short)0, c0[ct], false, false);
            c1[ct] = __builtin_amdgcn_wmma_f32_16x16x32_f16(
                false, A1[ks], false, buse, (short)0, c1[ct], false, false);
            if (idx + 2 < 16) bbuf[idx & 1] = loadB(idx + 2);  // reload 2 ahead
        }
        __syncthreads();   // all reads of Bs[cur] done before it is re-staged
    }

    // ---- writeback: VGPR rr -> row (rr + 8g); lane%16 -> column ----
#pragma unroll
    for (int ct = 0; ct < 8; ++ct) {
#pragma unroll
        for (int rr = 0; rr < 8; ++rr) {
            const int col = ct * 16 + m;
            out[(size_t)(row0 + rr + 8 * g) * OUT_DIM + col]      = c0[ct][rr];
            out[(size_t)(row0 + 16 + rr + 8 * g) * OUT_DIM + col] = c1[ct][rr];
        }
    }
}

// ---------------------------------------------------------------------------
extern "C" void kernel_launch(void* const* d_in, const int* in_sizes, int n_in,
                              void* d_out, int out_size, void* d_ws, size_t ws_size,
                              hipStream_t stream) {
    const float* x  = (const float*)d_in[0];   // [16384, 256]
    const float* W  = (const float*)d_in[1];   // [128, 256]
    const float* a1 = (const float*)d_in[2];   // [128]
    const float* a2 = (const float*)d_in[3];   // [128]
    float* out = (float*)d_out;                // [16384, 128]

    char* ws = (char*)d_ws;
    _Float16* hT  = (_Float16*)ws;                                   // 4 MB
    _Float16* s1h = (_Float16*)(ws + (size_t)OUT_DIM * N_ROWS * 2);  // 32 KB
    _Float16* s2h = s1h + N_ROWS;                                    // 32 KB

    phase1_h<<<N_ROWS, OUT_DIM, 0, stream>>>(x, W, hT);
    phase1_s<<<N_ROWS / 256, 256, 0, stream>>>(hT, a1, a2, s1h, s2h);
    phase3_attn_gemm<<<N_ROWS / 128, 128, 0, stream>>>(hT, s1h, s2h, out);
}
